// Correlation_18124761989821
// MI455X (gfx1250) — matrix operand order repaired and checked by reading, not compiled, not run on hardware
//
#include <hip/hip_runtime.h>

typedef __attribute__((ext_vector_type(16))) _Float16 v16h;
typedef __attribute__((ext_vector_type(8)))  _Float16 v8h;
typedef __attribute__((ext_vector_type(8)))  float    v8f;
typedef unsigned int v4u __attribute__((ext_vector_type(4)));
typedef int          v8i __attribute__((ext_vector_type(8)));
typedef int          v4i __attribute__((ext_vector_type(4)));

#define NB   8
#define NC   256
#define NH   64
#define NW   96
#define ND   21                 // displacement grid width (441 channels)
#define CP   264                // LDS row pitch in halves: 528 B rows, 16B aligned
#define NTHREADS 192

// ---- workspace layout: f16, transposed + parity-permuted [b][y][j][c] ------
// j in [0,96): j<48 -> x = 2j (even), j>=48 -> x = 2(j-48)+1 (odd)
#define TPLANE ((size_t)NW * NC)                 // halves per (b,y) row
#define TSIZE  ((size_t)NB * NH * TPLANE)        // halves per tensor

// ---- main-kernel LDS: 2 x parity-permuted u-row buffers + out tile ---------
// even u -> row (u+20)/2 in [0,72); odd u -> 72 + (u+19)/2 in [72,144)
#define BROWS 144
#define NPADROWS 48
#define BUFH   (BROWS*CP)                             // halves per buffer
#define BUFB   (BROWS*CP*2)                           // bytes per buffer (76,032)
#define LDS_O_OFF (2*BUFB)                            // bytes
#define SMEM_MAIN (2*BUFB + ND*NW*4)                  // 160,128 B
#define SMEM_PREP (NW*CP*2)                           // 50,688 B

__device__ __forceinline__ int permA(int x) {         // x in [0,96)
  return (x >> 1) + (x & 1) * 48;
}

// ---- TDM: DMA 48 rows x 512 B (global, stride 512 B) -> LDS, 528 B pitch ---
// D# per CDNA5 ISA: count=1, type=2(image),
// data_size=1(2B), pad_enable, pad_interval=6 (128 dw), pad_amount=3 (4 dw),
// tensor_dim0=256, tensor_dim1=48, tile_dim0=256, tile_dim1=48, stride0=256.
__device__ __forceinline__ void tdm_rows48(const _Float16* src, unsigned lds_off)
{
  unsigned long long ga = (unsigned long long)src;
  v4u g0;
  g0[0] = 1u;                                            // count=1 (user mode)
  g0[1] = lds_off;                                       // lds_addr (bytes)
  g0[2] = (unsigned)ga;                                  // global_addr[31:0]
  g0[3] = (unsigned)((ga >> 32) & 0x01FFFFFFull) | (2u << 30); // [56:32]|type=2
  v8i g1;
  g1[0] = (int)((1u << 16) | (1u << 20) | (6u << 22) | (3u << 25));
  g1[1] = (int)(256u << 16);                             // tensor_dim0 = 256
  g1[2] = (int)(48u << 16);                              // tensor_dim1 = 48
  g1[3] = (int)(256u << 16);                             // tile_dim0 = 256
  g1[4] = 48;                                            // tile_dim1 = 48
  g1[5] = 256;                                           // tensor_dim0_stride
  g1[6] = 0;
  g1[7] = 0;
  v4i z4 = {};
  v8i z8 = {};
  // amdgpu-toolchain (clang-23) 6-arg form: (g0, g1, g2, g3, extra, cpol)
  __builtin_amdgcn_tensor_load_to_lds(g0, g1, z4, z4, z8, 0);
}

// ======================= prep: f32 NCHW -> f16 [b][y][j][c] =================
__global__ __launch_bounds__(NTHREADS)
void corr_prep(const float* __restrict__ in1, const float* __restrict__ in2,
               _Float16* __restrict__ t1, _Float16* __restrict__ t2)
{
  const int b   = (int)blockIdx.x >> 6;
  const int y   = (int)blockIdx.x & 63;
  const int tid = (int)threadIdx.x;
  extern __shared__ char smem[];
  _Float16* T = (_Float16*)smem;                      // [96][CP]

  const float* srcs[2] = { in1 + ((size_t)b * NC * NH + y) * NW,
                           in2 + ((size_t)b * NC * NH + y) * NW };
  _Float16*    dsts[2] = { t1 + ((size_t)b * NH + y) * TPLANE,
                           t2 + ((size_t)b * NH + y) * TPLANE };

  for (int s = 0; s < 2; ++s) {
    const float* src = srcs[s];
    for (int idx = tid; idx < NC * (NW / 4); idx += NTHREADS) {
      int c  = idx / (NW / 4);
      int xq = (idx % (NW / 4)) * 4;
      float4 v = *(const float4*)(src + (size_t)c * NH * NW + xq);
      T[permA(xq + 0) * CP + c] = (_Float16)v.x;
      T[permA(xq + 1) * CP + c] = (_Float16)v.y;
      T[permA(xq + 2) * CP + c] = (_Float16)v.z;
      T[permA(xq + 3) * CP + c] = (_Float16)v.w;
    }
    __syncthreads();
    _Float16* dst = dsts[s];
    for (int idx = tid; idx < NW * (NC / 8); idx += NTHREADS) {
      int row = idx >> 5;
      int col = (idx & 31) * 8;
      *(v8h*)(dst + (size_t)row * NC + col) = *(const v8h*)(T + row * CP + col);
    }
    __syncthreads();                      // T reused for second input
  }
}

// ======================= main: banded WMMA correlation ======================
__global__ __launch_bounds__(NTHREADS)
void corr_wmma(const _Float16* __restrict__ in1t,
               const _Float16* __restrict__ in2t,
               float* __restrict__ out)
{
  const int b    = (int)blockIdx.x >> 6;
  const int y    = (int)blockIdx.x & 63;
  const int tid  = (int)threadIdx.x;
  const int lane = tid & 31;
  const int wv   = tid >> 5;               // 0..5
  const int hl   = lane >> 4;              // lane half
  const int nn   = lane & 15;              // lane-in-half
  const int p    = wv / 3;                 // x-parity handled by this wave
  const int i0   = (wv % 3) * 16;          // M-tile start (parity index)

  extern __shared__ char smem[];
  _Float16* B0 = (_Float16*)smem;          // two [144][CP] buffers
  float*    Os = (float*)(smem + LDS_O_OFF); // block-shared [21][96]

  // ---- zero pad rows of BOTH buffers: prow in [0,10) u [58,82) u [130,144) -
  {
    const v8h z8 = {};
    for (int idx = tid; idx < 2 * NPADROWS * (CP / 8); idx += NTHREADS) {
      int q    = idx / (CP / 8);
      int col  = (idx % (CP / 8)) * 8;
      int buf  = q / NPADROWS;
      int row  = q % NPADROWS;
      int prow = (row < 10) ? row : ((row < 34) ? (row + 48) : (row + 96));
      *(v8h*)(B0 + buf * BUFH + prow * CP + col) = z8;
    }
  }

  // ---- resident A fragments straight from global (pre-transposed f16) ------
  union VA { v16h v; v8h h[2]; };
  const _Float16* arow =
      in1t + ((size_t)b * NH + y) * TPLANE + (size_t)(i0 + nn + p * 48) * NC;
  v16h afrag[8];
  #pragma unroll
  for (int kk = 0; kk < 8; ++kk) {
    VA t;
    t.h[0] = *(const v8h*)(arow + kk * 32 + 16 * hl);
    t.h[1] = *(const v8h*)(arow + kk * 32 + 16 * hl + 8);
    afrag[kk] = t.v;
  }

  // per-lane B fragment offsets (halves, within a buffer); nt=2/p=1 over-reads
  // never-harvested columns, staying inside the 160,128 B allocation.
  int boff[3];
  #pragma unroll
  for (int nt = 0; nt < 3; ++nt)
    boff[nt] = (i0 + 16 * nt + nn + p * 72) * CP + 16 * hl;

  // ---- TDM prologue: DMA dy=0's row into buffer 0 (async) ------------------
  {
    const int r0 = y - 20;                 // r0 <= 43 < NH always
    if (wv == 0 && r0 >= 0) {
      const _Float16* src = in2t + ((size_t)b * NH + r0) * TPLANE;
      tdm_rows48(src,           10 * 528); // even-x rows -> prow 10..57
      tdm_rows48(src + 48 * NC, 82 * 528); // odd-x rows  -> prow 82..129
    }
  }

  // ---------------- loop over dy ----------------
  for (int dy = 0; dy < ND; ++dy) {
    const int  r     = y + 2 * dy - 20;
    const bool valid = (r >= 0) && (r < NH);
    const int  cur   = dy & 1;

    __builtin_amdgcn_s_wait_tensorcnt(0);  // wave0: DMA for dy complete
    __syncthreads();                       // publish buf[cur]; prev reads done

    // issue next dy's DMA into the other buffer, overlapped with compute
    {
      const int rn = r + 2;
      if (wv == 0 && dy + 1 < ND && rn >= 0 && rn < NH) {
        const _Float16* src = in2t + ((size_t)b * NH + rn) * TPLANE;
        const unsigned base = (unsigned)((dy + 1) & 1) * (unsigned)BUFB;
        tdm_rows48(src,           base + 10 * 528);
        tdm_rows48(src + 48 * NC, base + 82 * 528);
      }
    }

    if (valid) {
      const _Float16* bcur = B0 + (size_t)cur * BUFH;
      // dense banded GEMM on the parity grid: dx = 16*nt + n - m
      v8f  acc[3] = {};
      v16h bfrag[2][3];
      #pragma unroll
      for (int nt = 0; nt < 3; ++nt) {
        VA t;
        t.h[0] = *(const v8h*)(bcur + boff[nt]);
        t.h[1] = *(const v8h*)(bcur + boff[nt] + 8);
        bfrag[0][nt] = t.v;
      }
      #pragma unroll
      for (int kk = 0; kk < 8; ++kk) {
        const int c2 = kk & 1, nxt = c2 ^ 1;
        if (kk < 7) {
          #pragma unroll
          for (int nt = 0; nt < 3; ++nt) {
            const _Float16* bp = bcur + boff[nt] + (kk + 1) * 32;
            VA t;
            t.h[0] = *(const v8h*)(bp);
            t.h[1] = *(const v8h*)(bp + 8);
            bfrag[nxt][nt] = t.v;
          }
        }
        #pragma unroll
        for (int nt = 0; nt < 3; ++nt)
          acc[nt] = __builtin_amdgcn_wmma_f32_16x16x32_f16(
              false, afrag[kk], false, bfrag[c2][nt], (short)0, acc[nt],
              false, false);
      }

      // harvest: each (m, dx) maps to exactly one (nt, n): dx = 16*nt + n - m
      #pragma unroll
      for (int nt = 0; nt < 3; ++nt) {
        #pragma unroll
        for (int vi = 0; vi < 8; ++vi) {
          const int mm = vi + 8 * hl;
          const int dx = 16 * nt + nn - mm;
          if (dx >= 0 && dx <= 20)
            Os[dx * NW + 2 * (i0 + mm) + p] = acc[nt][vi];
        }
      }
    }
    __syncthreads();                       // Os complete before block-wide store

    // fully coalesced store of the 21x96 tile (one dy, all dx, all x)
    const size_t obase =
        (((size_t)b * (ND * ND) + (size_t)dy * ND) * NH + y) * NW;
    for (int idx = tid; idx < ND * NW; idx += NTHREADS) {
      int dxr = idx / NW;
      int xx  = idx - dxr * NW;
      float val = valid ? Os[idx] * (1.0f / 256.0f) : 0.0f;
      out[obase + (size_t)dxr * NH * NW + xx] = val;
    }
  }
}

extern "C" void kernel_launch(void* const* d_in, const int* in_sizes, int n_in,
                              void* d_out, int out_size, void* d_ws, size_t ws_size,
                              hipStream_t stream) {
  (void)in_sizes; (void)n_in; (void)out_size; (void)ws_size;
  const float* in1 = (const float*)d_in[0];
  const float* in2 = (const float*)d_in[1];
  float* out = (float*)d_out;

  // workspace: two f16 transposed tensors, 25.2 MB each (50.4 MB total)
  _Float16* t1 = (_Float16*)d_ws;
  _Float16* t2 = t1 + TSIZE;

  (void)hipFuncSetAttribute((const void*)corr_prep,
                            hipFuncAttributeMaxDynamicSharedMemorySize,
                            SMEM_PREP);
  (void)hipFuncSetAttribute((const void*)corr_wmma,
                            hipFuncAttributeMaxDynamicSharedMemorySize,
                            SMEM_MAIN);

  dim3 grid(NB * NH);       // 512 blocks: one per (b, y)
  dim3 block(NTHREADS);

  hipLaunchKernelGGL(corr_prep, grid, block, SMEM_PREP, stream,
                     in1, in2, t1, t2);
  hipLaunchKernelGGL(corr_wmma, grid, block, SMEM_MAIN, stream,
                     t1, t2, out);
}
